// CSAM_4664334483780
// MI455X (gfx1250) — matrix operand order repaired and checked by reading, not compiled, not run on hardware
//
#include <hip/hip_runtime.h>

typedef __bf16 bf16_t;
typedef __attribute__((ext_vector_type(4)))  float   v4f;
typedef __attribute__((ext_vector_type(4)))  __bf16  v4bf;
typedef __attribute__((ext_vector_type(8)))  __bf16  v8bf;
typedef __attribute__((ext_vector_type(16))) __bf16  v16bf;
typedef __attribute__((ext_vector_type(8)))  float   v8f;

#define C_DIM 512
#define N_DIM 4096
#define LDSS  40   // padded LDS row stride (bf16): 80B, multiple of 16B for b128 loads

// A fragment (16x32 bf16, CDNA5 layout): lane<16 -> row m, K = {0..7, 16..23};
// lane>=16 -> row m, K = {8..15, 24..31}. Two contiguous 8-elem halves -> ds_load_b128.
static __device__ __forceinline__ v16bf frag_a(const bf16_t* row, int hi) {
  union { v16bf f; v8bf h[2]; } u;
  u.h[0] = *(const v8bf*)(row + hi * 8);
  u.h[1] = *(const v8bf*)(row + hi * 8 + 16);
  return u.f;
}
// B fragment (32x16 bf16): lane<16 -> K = {0..7, 16..23}; lane>=16 -> K = {8..15, 24..31}, N = lane&15.
// B staged in LDS as [n][k], so two contiguous 8-elem halves -> two ds_load_b128.
static __device__ __forceinline__ v16bf frag_b(const bf16_t* row, int hi) {
  union { v16bf f; v8bf h[2]; } u;
  u.h[0] = *(const v8bf*)(row + hi * 8);
  u.h[1] = *(const v8bf*)(row + hi * 8 + 16);
  return u.f;
}

static __device__ __forceinline__ v4bf cvt4(v4f v) {
  v4bf r;
  r.x = (bf16_t)v.x; r.y = (bf16_t)v.y; r.z = (bf16_t)v.z; r.w = (bf16_t)v.w;
  return r;
}

#define WMMA_BF16(A, B, CC) \
  __builtin_amdgcn_wmma_f32_16x16x32_bf16(false, (A), false, (B), (short)0, (CC), false, false)

// ---------------- Kernel 1: energy[b] = q[b] * q[b]^T  (512x512 per batch) ----------------
// 128x128 tile / workgroup, 8 waves each 32x64 (8 WMMA per K-step), double-buffered LDS.
__global__ __launch_bounds__(256) void csam_gram(const float* __restrict__ x,
                                                 float* __restrict__ energy) {
  __shared__ __align__(16) bf16_t Ab[2][128][LDSS];
  __shared__ __align__(16) bf16_t Bb[2][128][LDSS];
  const int b    = blockIdx.z;
  const int row0 = blockIdx.y * 128;
  const int col0 = blockIdx.x * 128;
  const float* q = x + (size_t)b * C_DIM * N_DIM;

  const int tid  = threadIdx.x;
  const int lane = tid & 31, wave = tid >> 5;
  const int m    = lane & 15, hi = lane >> 4;
  const int wm   = wave >> 1, wn = wave & 1;   // 4x2 waves, each 32(M) x 64(N)

  // fill indexing: 128 rows x 32 K per operand, float4 chunks, 4 chunks each
  const int rF = tid >> 3;            // 0..31, + i*32
  const int kF = (tid & 7) * 4;       // 0..28

  v8f acc[2][4] = {};
  v4f areg[4], breg[4];

  // preload tile 0
#pragma unroll
  for (int i = 0; i < 4; ++i) {
    areg[i] = *(const v4f*)&q[(size_t)(row0 + rF + i * 32) * N_DIM + kF];
    breg[i] = *(const v4f*)&q[(size_t)(col0 + rF + i * 32) * N_DIM + kF];
  }
#pragma unroll
  for (int i = 0; i < 4; ++i) {
    *(v4bf*)&Ab[0][rF + i * 32][kF] = cvt4(areg[i]);
    *(v4bf*)&Bb[0][rF + i * 32][kF] = cvt4(breg[i]);
  }
  __syncthreads();

  const int S = N_DIM / 32;
  for (int s = 0; s < S; ++s) {
    const int p = s & 1;
    const bool more = (s + 1 < S);
    if (more) {
      const int k0 = (s + 1) * 32;
#pragma unroll
      for (int i = 0; i < 4; ++i) {
        areg[i] = *(const v4f*)&q[(size_t)(row0 + rF + i * 32) * N_DIM + k0 + kF];
        breg[i] = *(const v4f*)&q[(size_t)(col0 + rF + i * 32) * N_DIM + k0 + kF];
      }
    }
    // compute from buffer p
    const bf16_t* Abase = &Ab[p][wm * 32 + m][0];
    const bf16_t* Bbase = &Bb[p][wn * 64 + m][0];
    v16bf a0 = frag_a(Abase, hi);
    v16bf a1 = frag_a(Abase + 16 * LDSS, hi);
    v16bf bf[4];
#pragma unroll
    for (int j = 0; j < 4; ++j) bf[j] = frag_b(Bbase + j * 16 * LDSS, hi);
#pragma unroll
    for (int j = 0; j < 4; ++j) acc[0][j] = WMMA_BF16(a0, bf[j], acc[0][j]);
#pragma unroll
    for (int j = 0; j < 4; ++j) acc[1][j] = WMMA_BF16(a1, bf[j], acc[1][j]);
    if (more) {
#pragma unroll
      for (int i = 0; i < 4; ++i) {
        *(v4bf*)&Ab[1 - p][rF + i * 32][kF] = cvt4(areg[i]);
        *(v4bf*)&Bb[1 - p][rF + i * 32][kF] = cvt4(breg[i]);
      }
    }
    __syncthreads();
  }

  float* e = energy + (size_t)b * C_DIM * C_DIM;
#pragma unroll
  for (int i = 0; i < 2; ++i)
#pragma unroll
    for (int j = 0; j < 4; ++j)
#pragma unroll
      for (int r = 0; r < 8; ++r) {
        int M0 = row0 + wm * 32 + i * 16 + r + hi * 8;
        int N0 = col0 + wn * 64 + j * 16 + m;
        e[(size_t)M0 * C_DIM + N0] = acc[i][j][r];
      }
}

// ---- Kernel 2: attention = softmax(rowmax - e) == exp(rowmin - e)/sum, in place ----
__global__ __launch_bounds__(256) void csam_softmax(float* __restrict__ energy) {
  const int lane = threadIdx.x & 31;
  const int wave = threadIdx.x >> 5;
  const int row  = blockIdx.x * 8 + wave;      // 16*512 = 8192 rows, one wave each
  float* e = energy + (size_t)row * C_DIM;

  float v[16];
  float mn = 3.4e38f;
#pragma unroll
  for (int t = 0; t < 16; ++t) {
    v[t] = e[lane + 32 * t];
    mn = fminf(mn, v[t]);
  }
#pragma unroll
  for (int s = 16; s > 0; s >>= 1) mn = fminf(mn, __shfl_xor(mn, s, 32));
  float sum = 0.f;
#pragma unroll
  for (int t = 0; t < 16; ++t) {
    v[t] = __expf(mn - v[t]);
    sum += v[t];
  }
#pragma unroll
  for (int s = 16; s > 0; s >>= 1) sum += __shfl_xor(sum, s, 32);
  const float inv = 1.0f / sum;
#pragma unroll
  for (int t = 0; t < 16; ++t) e[lane + 32 * t] = v[t] * inv;
}

// ---- Kernel 3: out = gamma * (attn @ q) + x  (512x4096 per batch) ----
__global__ __launch_bounds__(256) void csam_av(const float* __restrict__ attn,
                                               const float* __restrict__ x,
                                               const float* __restrict__ gamma,
                                               float* __restrict__ out) {
  __shared__ __align__(16) bf16_t Ab[2][128][LDSS];
  __shared__ __align__(16) bf16_t Bb[2][128][LDSS];   // transposed: Bb[.][n][k]
  const int b    = blockIdx.z;
  const int row0 = blockIdx.y * 128;                  // channel rows
  const int n0   = blockIdx.x * 128;                  // spatial cols
  const float* A = attn + (size_t)b * C_DIM * C_DIM;
  const float* q = x + (size_t)b * C_DIM * N_DIM;

  const int tid  = threadIdx.x;
  const int lane = tid & 31, wave = tid >> 5;
  const int m    = lane & 15, hi = lane >> 4;
  const int wm   = wave >> 1, wn = wave & 1;

  // A fill: 128 rows x 32 K
  const int rF = tid >> 3;
  const int kF = (tid & 7) * 4;
  // B fill: 32 K-rows x 128 n-cols, transposed into LDS
  const int kB = tid >> 5;            // 0..7, + i*8
  const int nB = (tid & 31) * 4;      // 0..124

  v8f acc[2][4] = {};
  v4f areg[4], breg[4];

#pragma unroll
  for (int i = 0; i < 4; ++i) {
    areg[i] = *(const v4f*)&A[(size_t)(row0 + rF + i * 32) * C_DIM + kF];
    breg[i] = *(const v4f*)&q[(size_t)(kB + i * 8) * N_DIM + n0 + nB];
  }
#pragma unroll
  for (int i = 0; i < 4; ++i) {
    *(v4bf*)&Ab[0][rF + i * 32][kF] = cvt4(areg[i]);
#pragma unroll
    for (int j = 0; j < 4; ++j) Bb[0][nB + j][kB + i * 8] = (bf16_t)breg[i][j];
  }
  __syncthreads();

  const int S = C_DIM / 32;
  for (int s = 0; s < S; ++s) {
    const int p = s & 1;
    const bool more = (s + 1 < S);
    if (more) {
      const int k0 = (s + 1) * 32;
#pragma unroll
      for (int i = 0; i < 4; ++i) {
        areg[i] = *(const v4f*)&A[(size_t)(row0 + rF + i * 32) * C_DIM + k0 + kF];
        breg[i] = *(const v4f*)&q[(size_t)(k0 + kB + i * 8) * N_DIM + n0 + nB];
      }
    }
    const bf16_t* Abase = &Ab[p][wm * 32 + m][0];
    const bf16_t* Bbase = &Bb[p][wn * 64 + m][0];
    v16bf a0 = frag_a(Abase, hi);
    v16bf a1 = frag_a(Abase + 16 * LDSS, hi);
    v16bf bf[4];
#pragma unroll
    for (int j = 0; j < 4; ++j) bf[j] = frag_b(Bbase + j * 16 * LDSS, hi);
#pragma unroll
    for (int j = 0; j < 4; ++j) acc[0][j] = WMMA_BF16(a0, bf[j], acc[0][j]);
#pragma unroll
    for (int j = 0; j < 4; ++j) acc[1][j] = WMMA_BF16(a1, bf[j], acc[1][j]);
    if (more) {
#pragma unroll
      for (int i = 0; i < 4; ++i) {
        *(v4bf*)&Ab[1 - p][rF + i * 32][kF] = cvt4(areg[i]);
#pragma unroll
        for (int j = 0; j < 4; ++j) Bb[1 - p][nB + j][kB + i * 8] = (bf16_t)breg[i][j];
      }
    }
    __syncthreads();
  }

  const float g = gamma[0];
  float* o = out + (size_t)b * C_DIM * N_DIM;
#pragma unroll
  for (int i = 0; i < 2; ++i)
#pragma unroll
    for (int j = 0; j < 4; ++j)
#pragma unroll
      for (int r = 0; r < 8; ++r) {
        int M0 = row0 + wm * 32 + i * 16 + r + hi * 8;
        int N0 = n0 + wn * 64 + j * 16 + m;
        size_t idx = (size_t)M0 * N_DIM + N0;
        o[idx] = g * acc[i][j][r] + q[idx];
      }
}

extern "C" void kernel_launch(void* const* d_in, const int* in_sizes, int n_in,
                              void* d_out, int out_size, void* d_ws, size_t ws_size,
                              hipStream_t stream) {
  (void)in_sizes; (void)n_in; (void)out_size; (void)ws_size;
  const float* x     = (const float*)d_in[0];
  const float* gamma = (const float*)d_in[1];
  float* out         = (float*)d_out;
  float* energy      = (float*)d_ws;       // 16 * 512 * 512 * 4B = 16 MB scratch

  // energy = q q^T
  csam_gram<<<dim3(C_DIM / 128, C_DIM / 128, 16), 256, 0, stream>>>(x, energy);
  // attention = softmax(rowmax - energy), in place
  csam_softmax<<<dim3((16 * C_DIM) / 8), 256, 0, stream>>>(energy);
  // out = gamma * (attention @ q) + x
  csam_av<<<dim3(N_DIM / 128, C_DIM / 128, 16), 256, 0, stream>>>(energy, x, gamma, out);
}
